// Decoder_6528350290405
// MI455X (gfx1250) — matrix operand order repaired and checked by reading, not compiled, not run on hardware
//
#include <hip/hip_runtime.h>

#define B_TOT 64
#define LSEQ  128
#define EDIM  256
#define HDIM  256

typedef __attribute__((ext_vector_type(16))) _Float16 v16h;
typedef __attribute__((ext_vector_type(8)))  float    v8f;

__device__ __forceinline__ v8f wmma16(v16h a, v16h b, v8f c) {
  // 8 args: (neg_a, A, neg_b, B, c_mod, C, reuse_a, reuse_b)
  return __builtin_amdgcn_wmma_f32_16x16x32_f16(false, a, false, b, (short)0, c, false, false);
}

// A fragment (16x32 f16) from row-major LDS (stride in halves). Per-lane data is
// two contiguous 16B runs -> 2x ds_load_b128.
// ISA 7.12.2: lanes 0-15 M=lane K={0..7,16..23}; lanes 16-31 M=lane-16 K={8..15,24..31}
__device__ __forceinline__ v16h load_a_lds(const _Float16* A, int strideH, int k0, int lane) {
  const int m  = lane & 15;
  const int kb = (lane >> 4) << 3;   // 0 or 8
  const uint4* row = (const uint4*)(A + m * strideH + k0 + kb);
  union { v16h v; uint4 q[2]; } r;
  r.q[0] = row[0];   // halves k0+kb .. +7
  r.q[1] = row[2];   // halves k0+kb+16 .. +23
  return r.v;
}

// B fragment from fragment-major packed buffer: p = fragBase + lane*2 (uint4
// units); one fragment = 32 lanes x 32B contiguous -> 2x global_load_b128.
__device__ __forceinline__ v16h load_b_frag(const uint4* __restrict__ p) {
  union { v16h v; uint4 q[2]; } r;
  r.q[0] = p[0];
  r.q[1] = p[1];
  return r.v;
}

__device__ __forceinline__ float fast_tanh(float x) {
#if __has_builtin(__builtin_amdgcn_tanhf)
  return __builtin_amdgcn_tanhf(x);        // v_tanh_f32 (native CDNA5 trans op)
#else
  float e = __expf(2.f * x);               // v_exp_f32 path; saturates correctly
  return 1.f - 2.f / (e + 1.f);
#endif
}
__device__ __forceinline__ float sigm(float x) { return 1.f / (1.f + __expf(-x)); }

// ---- prep: f32 [K][N] weights -> f16 fragment-major WMMA-B layout ------------
// dword gid = ((ntile*ksteps + kstep)*32 + lane)*8 + i
// lane: n = ntile*16 + (lane&15); kpair = kstep*16 + ((lane>=16)?4:0) + (i&3) + ((i>>2)<<3)
__global__ void pack_w_frag(const float* __restrict__ W, unsigned* __restrict__ P,
                            int K, int N) {
  int gid = blockIdx.x * blockDim.x + threadIdx.x;
  if (gid >= (K * N) >> 1) return;
  const int ksteps = K >> 5;
  int i    = gid & 7;
  int lane = (gid >> 3) & 31;
  int f    = gid >> 8;
  int ntile = f / ksteps;
  int kstep = f - ntile * ksteps;
  int n  = ntile * 16 + (lane & 15);
  int k  = (kstep * 16 + ((lane >> 4) << 2) + (i & 3) + ((i >> 2) << 3)) << 1;
  union { _Float16 h[2]; unsigned u; } pk;
  pk.h[0] = (_Float16)W[(size_t)k * N + n];
  pk.h[1] = (_Float16)W[(size_t)(k + 1) * N + n];
  P[gid] = pk.u;
}

// ---- prep: ctx[b][l][k] = sum_h context[b][l][h] * Wc[h][k] + bc[k] (f32) ----
__global__ void ctx_kernel(const float* __restrict__ ctxIn, const float* __restrict__ Wc,
                           const float* __restrict__ bc, float* __restrict__ ctxOut) {
  int gid = blockIdx.x * blockDim.x + threadIdx.x;  // B*L*H
  int k  = gid & (HDIM - 1);
  int bl = gid >> 8;
  const float* row = ctxIn + (size_t)bl * HDIM;
  float s = bc[k];
#pragma unroll 4
  for (int h = 0; h < HDIM; ++h) s += row[h] * Wc[h * HDIM + k];
  ctxOut[gid] = s;
}

// ---- persistent decoder: 1 block = 16 batch rows, full 128-step scan ---------
__global__ __launch_bounds__(512) void decoder_kernel(
    const float* __restrict__ emb,  const float* __restrict__ dec0,
    const float* __restrict__ h0,   const float* __restrict__ c0,
    const float* __restrict__ bi,   const float* __restrict__ bh,
    const float* __restrict__ ba,   const float* __restrict__ Vw,
    const float* __restrict__ bo,   const float* __restrict__ ctx,
    const uint4* __restrict__ WiF, const uint4* __restrict__ WhF,
    const uint4* __restrict__ WaF, const uint4* __restrict__ WoF,
    float* __restrict__ outAlpha, float* __restrict__ outPtr,
    float* __restrict__ outH, float* __restrict__ outC) {
  extern __shared__ char smem[];
  _Float16* xf16 = (_Float16*)smem;        // [16][256] dec_in (f16 A-matrix)
  _Float16* hf16 = xf16 + 16 * 256;        // [16][256] h_t    (f16 A-matrix)
  _Float16* hcat = hf16 + 16 * 256;        // [16][512] [hid | h_lstm] f16
  float* inp   = (float*)(hcat + 16 * 512);// [16][256] attention query
  float* attb  = inp + 16 * 256;           // [16][128] att scores -> alpha
  float* maskb = attb + 16 * 128;          // [16][128]
  int*   headb = (int*)(maskb + 16 * 128); // [16][128]
  float* biasg = (float*)(headb + 16 * 128);  // [1024] bi+bh
  float* baL   = biasg + 1024;             // [256]
  float* boL   = baL + 256;                // [256]
  float* VL    = boL + 256;                // [256]
  int*   curb  = (int*)(VL + 256);         // [16]
  int*   idxb  = curb + 16;                // [16]

  const int tid   = threadIdx.x;
  const int lane  = tid & 31;
  const int wv    = tid >> 5;         // wave owns output cols [16wv,16wv+16)
  const int b0    = blockIdx.x << 4;  // 16 batch rows per block
  const int hcol0 = wv << 4;
  const int mb    = (lane >> 4) << 3; // C/D frag: row = mb + r
  const int nn    = lane & 15;        // C/D frag: col = n0 + nn

  // per-wave fragment bases (uint4 units; one fragment = 64 uint4)
  // Wi/Wh: 64 ntiles x 8 ksteps; gate g tile = g*16+wv  -> gate stride 16*8*64 = 8192
  const uint4* WiBase = WiF + (size_t)(wv * 8) * 64 + lane * 2;
  const uint4* WhBase = WhF + (size_t)(wv * 8) * 64 + lane * 2;
  const uint4* WaBase = WaF + (size_t)(wv * 8) * 64 + lane * 2;   // 16 ntiles x 8 ksteps
  const uint4* WoBase = WoF + (size_t)(wv * 16) * 64 + lane * 2;  // 16 ntiles x 16 ksteps

  // ---- init state ----
  for (int p = tid; p < 1024; p += 512) biasg[p] = bi[p] + bh[p];
  for (int p = tid; p < 256; p += 512) { baL[p] = ba[p]; boL[p] = bo[p]; VL[p] = Vw[p]; }
  for (int p = tid; p < 16 * 256; p += 512) {
    int b = p >> 8, e = p & 255;
    xf16[p] = (_Float16)dec0[(size_t)(b0 + b) * 256 + e];
    hf16[p] = (_Float16)h0[(size_t)(b0 + b) * 256 + e];
  }
  for (int p = tid; p < 16 * 128; p += 512) { maskb[p] = 1.f; headb[p] = p & 127; }
  if (tid < 16) curb[tid] = 0;
  v8f cfrag;
#pragma unroll
  for (int r = 0; r < 8; ++r) cfrag[r] = c0[(size_t)(b0 + mb + r) * 256 + hcol0 + nn];
  __syncthreads();

  for (int t = 0; t < LSEQ; ++t) {
    // ---- 1. LSTM gates: x@Wi + h@Wh (+biases); rolled k-loops, b128 frags ----
    v8f gi = {0,0,0,0,0,0,0,0}, gf = {0,0,0,0,0,0,0,0};
    v8f gg = {0,0,0,0,0,0,0,0}, go = {0,0,0,0,0,0,0,0};
    {
      const uint4* p = WiBase;
#pragma unroll 1
      for (int k0 = 0; k0 < 256; k0 += 32) {
        v16h ax = load_a_lds(xf16, 256, k0, lane);
        gi = wmma16(ax, load_b_frag(p),             gi);
        gf = wmma16(ax, load_b_frag(p + 1 * 8192),  gf);
        gg = wmma16(ax, load_b_frag(p + 2 * 8192),  gg);
        go = wmma16(ax, load_b_frag(p + 3 * 8192),  go);
        p += 64;
      }
    }
    {
      const uint4* p = WhBase;
#pragma unroll 1
      for (int k0 = 0; k0 < 256; k0 += 32) {
        v16h ah = load_a_lds(hf16, 256, k0, lane);
        gi = wmma16(ah, load_b_frag(p),             gi);
        gf = wmma16(ah, load_b_frag(p + 1 * 8192),  gf);
        gg = wmma16(ah, load_b_frag(p + 2 * 8192),  gg);
        go = wmma16(ah, load_b_frag(p + 3 * 8192),  go);
        p += 64;
      }
    }
    {
      const int col = hcol0 + nn;
      const float bii = biasg[col], bff = biasg[256 + col];
      const float bgg = biasg[512 + col], boo = biasg[768 + col];
#pragma unroll
      for (int r = 0; r < 8; ++r) {
        float igt = sigm(gi[r] + bii);
        float fgt = sigm(gf[r] + bff);
        float ggt = fast_tanh(gg[r] + bgg);
        float ogt = sigm(go[r] + boo);
        float cn  = fgt * cfrag[r] + igt * ggt;   // cell state stays in registers
        cfrag[r]  = cn;
        float hl  = ogt * fast_tanh(cn);
        hcat[(mb + r) * 512 + 256 + col] = (_Float16)hl;  // h_lstm half of concat
      }
    }
    __syncthreads();

    // ---- 2. inp = h_lstm @ Wa + ba ----
    {
      v8f pa = {0,0,0,0,0,0,0,0};
      const uint4* p = WaBase;
#pragma unroll 1
      for (int k0 = 0; k0 < 256; k0 += 32) {
        v16h ah = load_a_lds(hcat + 256, 512, k0, lane);
        pa = wmma16(ah, load_b_frag(p), pa);
        p += 64;
      }
      const int n = hcol0 + nn;
      const float bav = baL[n];
#pragma unroll
      for (int r = 0; r < 8; ++r) inp[(mb + r) * 256 + n] = pa[r] + bav;
    }
    __syncthreads();

    // ---- 3. att[b][l] = sum_h V[h]*tanh(inp[b][h] + ctx[b][l][h]) ----
#pragma unroll
    for (int j = 0; j < 4; ++j) {
      int p = tid + j * 512;                     // 2048 (b,l) pairs
      int b = p >> 7, l = p & 127;
      const float4* cr = (const float4*)(ctx + (((size_t)(b0 + b) * 128 + l) << 8));
      const float4* ir = (const float4*)(inp + (b << 8));
      const float4* vr = (const float4*)VL;
      float s = 0.f;
#pragma unroll 1
      for (int h4 = 0; h4 < 64; ++h4) {
        float4 cv = cr[h4], iv = ir[h4], vv = vr[h4];
        s += vv.x * fast_tanh(iv.x + cv.x) + vv.y * fast_tanh(iv.y + cv.y)
           + vv.z * fast_tanh(iv.z + cv.z) + vv.w * fast_tanh(iv.w + cv.w);
      }
      attb[p] = (maskb[p] > 0.f) ? s : -1.0e30f;
    }
    __syncthreads();

    // ---- 4. softmax over l, one wave per batch row; emit alpha ----
    {
      const int b = wv;
      float* row = attb + b * 128;
      float v[4], mx = -1.0e30f;
#pragma unroll
      for (int j = 0; j < 4; ++j) { v[j] = row[lane + j * 32]; mx = fmaxf(mx, v[j]); }
      for (int off = 16; off; off >>= 1) mx = fmaxf(mx, __shfl_xor(mx, off, 32));
      float sum = 0.f;
#pragma unroll
      for (int j = 0; j < 4; ++j) { v[j] = __expf(v[j] - mx); sum += v[j]; }
      for (int off = 16; off; off >>= 1) sum += __shfl_xor(sum, off, 32);
      const float inv = 1.f / sum;
#pragma unroll
      for (int j = 0; j < 4; ++j) {
        float a = v[j] * inv;
        row[lane + j * 32] = a;
        outAlpha[((size_t)(b0 + b) * 128 + t) * 128 + lane + j * 32] = a;
      }
    }
    __syncthreads();

    // ---- 5. hid[b][h] = sum_l ctx[b][l][h]*alpha[b][l]  -> hcat[:,0:256] ----
#pragma unroll
    for (int j = 0; j < 8; ++j) {
      int p = tid + j * 512;                     // 4096 (b,h) pairs
      int b = p >> 8, h = p & 255;
      const float* cb = ctx + (size_t)(b0 + b) * (128 * 256) + h;
      const float* ar = attb + b * 128;
      float s = 0.f;
#pragma unroll 4
      for (int l = 0; l < 128; ++l) s += cb[(size_t)l * 256] * ar[l];
      hcat[b * 512 + h] = (_Float16)s;
    }
    __syncthreads();

    // ---- 6. h_t = tanh([hid | h_lstm] @ Wo + bo) ----
    {
      v8f po = {0,0,0,0,0,0,0,0};
      const uint4* p = WoBase;
#pragma unroll 1
      for (int k0 = 0; k0 < 512; k0 += 32) {
        v16h ah = load_a_lds(hcat, 512, k0, lane);
        po = wmma16(ah, load_b_frag(p), po);
        p += 64;
      }
      const int n = hcol0 + nn;
      const float bov = boL[n];
#pragma unroll
      for (int r = 0; r < 8; ++r) {
        float hv = fast_tanh(po[r] + bov);
        hf16[(mb + r) * 256 + n] = (_Float16)hv;
        if (t == LSEQ - 1) outH[(size_t)(b0 + mb + r) * 256 + n] = hv;
      }
    }

    // ---- 7. greedy selection under temp mask, one wave per row ----
    {
      const int b = wv;
      const int cu = curb[b];
      const int chain = headb[b * 128 + cu];
      const float* ar = attb + b * 128;
      const float* mr = maskb + b * 128;
      float bv = -1.f; int bidx = 127;
      for (int l = lane; l < 128; l += 32) {
        float tm = mr[l];
        if (l == cu) tm = 0.f;
        if ((t != LSEQ - 1) && (l == chain)) tm = 0.f;
        float vv = ar[l] * tm;
        if (vv > bv || (vv == bv && l < bidx)) { bv = vv; bidx = l; }
      }
      for (int off = 16; off; off >>= 1) {
        float ov = __shfl_xor(bv, off, 32);
        int   oi = __shfl_xor(bidx, off, 32);
        if (ov > bv || (ov == bv && oi < bidx)) { bv = ov; bidx = oi; }
      }
      if (lane == 0) {
        idxb[b] = bidx;
        headb[b * 128 + bidx] = chain;
        maskb[b * 128 + bidx] = 0.f;
        curb[b] = bidx;
        outPtr[(size_t)(b0 + b) * 128 + t] = (float)bidx;
      }
    }
    __syncthreads();

    // ---- 8. dec_in = embedded_inputs[b, idx] -> xf16 ----
#pragma unroll
    for (int j = 0; j < 8; ++j) {
      int p = tid + j * 512;
      int b = p >> 8, e = p & 255;
      int ix = idxb[b];
      xf16[p] = (_Float16)emb[((size_t)(b0 + b) * 128 + ix) * 256 + e];
    }
    __syncthreads();
  }

  // ---- final cell state cN ----
#pragma unroll
  for (int r = 0; r < 8; ++r)
    outC[(size_t)(b0 + mb + r) * 256 + hcol0 + nn] = cfrag[r];
}

extern "C" void kernel_launch(void* const* d_in, const int* in_sizes, int n_in,
                              void* d_out, int out_size, void* d_ws, size_t ws_size,
                              hipStream_t stream) {
  (void)in_sizes; (void)n_in; (void)out_size; (void)ws_size;
  const float* emb  = (const float*)d_in[0];
  const float* dec0 = (const float*)d_in[1];
  const float* h0   = (const float*)d_in[2];
  const float* c0   = (const float*)d_in[3];
  const float* ctxI = (const float*)d_in[4];
  const float* Wi   = (const float*)d_in[5];
  const float* bi   = (const float*)d_in[6];
  const float* Wh   = (const float*)d_in[7];
  const float* bh   = (const float*)d_in[8];
  const float* Wa   = (const float*)d_in[9];
  const float* ba   = (const float*)d_in[10];
  const float* Wc   = (const float*)d_in[11];
  const float* bc   = (const float*)d_in[12];
  const float* Vw   = (const float*)d_in[13];
  const float* Wo   = (const float*)d_in[14];
  const float* bo   = (const float*)d_in[15];

  // workspace layout
  char* ws = (char*)d_ws;
  float*    ctxW = (float*)ws;                  // 64*128*256 f32 = 8 MB
  unsigned* WiP  = (unsigned*)(ws + 8388608);   // 256x1024 f16 frag-major (512 KB)
  unsigned* WhP  = (unsigned*)(ws + 8388608 + 524288);
  unsigned* WaP  = (unsigned*)(ws + 8388608 + 1048576);           // 128 KB
  unsigned* WoP  = (unsigned*)(ws + 8388608 + 1048576 + 131072);  // 256 KB

  float* out      = (float*)d_out;
  float* outAlpha = out;                              // [64][128][128]
  float* outPtr   = out + 64 * 128 * 128;             // [64][128]
  float* outH     = outPtr + 64 * 128;                // [64][256]
  float* outC     = outH + 64 * 256;                  // [64][256]

  // prep: pack weights to fragment-major f16 WMMA-B layout (dword counts = K*N/2)
  pack_w_frag<<<(131072 + 255) / 256, 256, 0, stream>>>(Wi, WiP, 256, 1024);
  pack_w_frag<<<(131072 + 255) / 256, 256, 0, stream>>>(Wh, WhP, 256, 1024);
  pack_w_frag<<<(32768 + 255) / 256, 256, 0, stream>>>(Wa, WaP, 256, 256);
  pack_w_frag<<<(65536 + 255) / 256, 256, 0, stream>>>(Wo, WoP, 512, 256);

  // prep: projected context (f32, reused by all 128 steps)
  ctx_kernel<<<(64 * 128 * 256) / 256, 256, 0, stream>>>(ctxI, Wc, bc, ctxW);

  // persistent decoder: 4 blocks x 16 batch rows, 512 threads (16 wave32s)
  const size_t smem = 16 * 256 * 2 * 2      // xf16, hf16
                    + 16 * 512 * 2          // hcat
                    + 16 * 256 * 4          // inp
                    + 16 * 128 * 4 * 3      // attb, maskb, headb
                    + 1024 * 4 + 3 * 256 * 4
                    + 2 * 16 * 4;           // curb, idxb
  decoder_kernel<<<4, 512, smem, stream>>>(emb, dec0, h0, c0, bi, bh, ba, Vw, bo,
                                           ctxW, (const uint4*)WiP, (const uint4*)WhP,
                                           (const uint4*)WaP, (const uint4*)WoP,
                                           outAlpha, outPtr, outH, outC);
}
